// CustomMultiHeadAttention_53386443489325
// MI455X (gfx1250) — matrix operand order repaired and checked by reading, not compiled, not run on hardware
//
#include <hip/hip_runtime.h>

#define DM 1024
#define NH 16
#define DH 64
#define BB 2
#define SS 2048

typedef __attribute__((ext_vector_type(16))) __bf16 v16bf;
typedef __attribute__((ext_vector_type(8)))  __bf16 v8bf;
typedef __attribute__((ext_vector_type(8)))  float  v8f;
typedef __attribute__((ext_vector_type(4)))  float  v4f;

// ---------------------------------------------------------------------------
// Fragment-linear offsets (ISA 7.12.2, wave32, bf16 16x16x32 WMMA):
// A 16x32:  lane = m + ((k>>3)&1)*16 ; elem e = ((k>>4)*8) + (k&7)
//   -> 8 consecutive k (fixed hi/group) are 8 consecutive fragment bytes*2
// B 32x16:  lane = n + ((k>>3)&1)*16 ; elem e = ((k>>4)*8) + (k&7)
//   -> same k order as A; 8 consecutive k (fixed hi/group) are contiguous
// C/D 16x16 f32: VGPR r: lanes0-15 -> M=r, lanes16-31 -> M=r+8; N = lane&15
// ---------------------------------------------------------------------------
__device__ __forceinline__ int a_frag_off(int mLocal, int k) {
  int hi = (k >> 3) & 1;
  int lane = mLocal + (hi << 4);
  int e = ((k >> 4) << 3) + (k & 7);
  return lane * 16 + e;
}

__device__ __forceinline__ void store8(__bf16* dst, v4f a, v4f b) {
  v8bf t;
  t[0] = (__bf16)a[0]; t[1] = (__bf16)a[1]; t[2] = (__bf16)a[2]; t[3] = (__bf16)a[3];
  t[4] = (__bf16)b[0]; t[5] = (__bf16)b[1]; t[6] = (__bf16)b[2]; t[7] = (__bf16)b[3];
  *reinterpret_cast<v8bf*>(dst) = t;          // one ds_store_b128
}

// gfx1250 async global->LDS copy, 16B per lane, tracked by ASYNCcnt.
__device__ __forceinline__ void async_copy16(const void* lds_dst, const void* gsrc) {
  unsigned loff = (unsigned)(uintptr_t)lds_dst;       // low 32 bits = LDS offset
  unsigned long long ga = (unsigned long long)(uintptr_t)gsrc;
  asm volatile("global_load_async_to_lds_b128 %0, %1, off"
               :: "v"(loff), "v"(ga) : "memory");
}

// ---------------------------------------------------------------------------
// Projection GEMM: Y = (X @ W^T + bias) * scale     X:[4096,1024] W:[N=1024,K]
// MODE 0: bf16 out [B,NH,S,DH]   (Q, K)
// MODE 2: bf16 out [B,NH,DH,S]   (V, transposed for contiguous attn staging)
// MODE 1: f32  out [M,N]         (final output)
// 128 threads / 4 waves; 64x64 tile; K-step 32; register-prefetch pipeline.
// ---------------------------------------------------------------------------
template <int MODE>
__global__ __launch_bounds__(128)
void proj_gemm_kernel(const float* __restrict__ A,
                      const float* __restrict__ W,
                      const float* __restrict__ bias,
                      unsigned short* __restrict__ out_bf,
                      float* __restrict__ out_f,
                      float scale)
{
  __shared__ __bf16 ldsA[4 * 512];
  __shared__ __bf16 ldsB[4 * 512];

  const int tid  = threadIdx.x;
  const int wave = tid >> 5, lane = tid & 31;
  const int row0 = blockIdx.x * 64;
  const int col0 = blockIdx.y * 64;

  v8f acc[4] = {};
  v4f ar[2][2], br[2][2];

  // chunk decode: g = c*128+tid ; idx = g>>2 (row/col), r = g&3
  auto load_tiles = [&](int kb) {
    #pragma unroll
    for (int c = 0; c < 2; ++c) {
      int g = c * 128 + tid;
      int m = g >> 2, r = g & 3, hi = r >> 1, lo = r & 1;
      // A chunk: k base = lo*16 + hi*8
      const float* pa = &A[(size_t)(row0 + m) * DM + kb + lo * 16 + hi * 8];
      ar[c][0] = *reinterpret_cast<const v4f*>(pa);
      ar[c][1] = *reinterpret_cast<const v4f*>(pa + 4);
      // B chunk: k base = lo*16 + hi*8
      const float* pb = &W[(size_t)(col0 + m) * DM + kb + lo * 16 + hi * 8];
      br[c][0] = *reinterpret_cast<const v4f*>(pb);
      br[c][1] = *reinterpret_cast<const v4f*>(pb + 4);
    }
  };

  load_tiles(0);

  for (int kb = 0; kb < DM; kb += 32) {
    // convert + swizzled store (one b128 per chunk)
    #pragma unroll
    for (int c = 0; c < 2; ++c) {
      int g = c * 128 + tid;
      int m = g >> 2, r = g & 3, hi = r >> 1, lo = r & 1;
      store8(&ldsA[((m >> 4) << 9) + ((m & 15) + (hi << 4)) * 16 + lo * 8],
             ar[c][0], ar[c][1]);
      store8(&ldsB[((m >> 4) << 9) + ((m & 15) + (hi << 4)) * 16 + lo * 8],
             br[c][0], br[c][1]);
    }
    __syncthreads();

    if (kb + 32 < DM) load_tiles(kb + 32);   // overlap next loads with WMMA

    v16bf a = *reinterpret_cast<const v16bf*>(&ldsA[wave << 9]);
    #pragma unroll
    for (int t = 0; t < 4; ++t) {
      v16bf b = *reinterpret_cast<const v16bf*>(&ldsB[t << 9]);
      acc[t] = __builtin_amdgcn_wmma_f32_16x16x32_bf16(
          false, a, false, b, (short)0, acc[t], false, false);
    }
    __syncthreads();
  }

  const int hi = lane >> 4, nl = lane & 15;
  #pragma unroll
  for (int t = 0; t < 4; ++t) {
    int gcol = col0 + t * 16 + nl;
    float bv = bias[gcol];
    #pragma unroll
    for (int i = 0; i < 8; ++i) {
      int grow = row0 + wave * 16 + i + 8 * hi;
      float v = (acc[t][i] + bv) * scale;
      if (MODE == 0) {
        int b = grow >> 11, s = grow & (SS - 1);
        int h = gcol >> 6,  dh = gcol & (DH - 1);
        ((__bf16*)out_bf)[(((size_t)(b * NH + h)) * SS + s) * DH + dh] = (__bf16)v;
      } else if (MODE == 2) {
        int b = grow >> 11, s = grow & (SS - 1);
        int h = gcol >> 6,  dh = gcol & (DH - 1);
        ((__bf16*)out_bf)[(((size_t)(b * NH + h)) * DH + dh) * SS + s] = (__bf16)v;
      } else {
        out_f[(size_t)grow * DM + gcol] = v;
      }
    }
  }
}

// ---------------------------------------------------------------------------
// Flash attention: WG = (64 q rows, one (b,h)); 4 waves, 16 q rows each.
// K/V tiles double-buffered in LDS, filled by async global->LDS b128 copies
// directly in WMMA B-fragment order; 64-key blocks with online softmax.
// Q bf16 [B,NH,S,DH] (pre-scaled), K bf16 [B,NH,S,DH], V bf16 [B,NH,DH,S].
// ---------------------------------------------------------------------------
__global__ __launch_bounds__(128)
void attention_kernel(const unsigned short* __restrict__ Qb,
                      const unsigned short* __restrict__ Kb,
                      const unsigned short* __restrict__ Vtb,
                      const float* __restrict__ pos_bias,  // [NH,S,S]
                      const int*   __restrict__ mask,      // [B,S,S]
                      float* __restrict__ ctx)             // [B*S, DM]
{
  __shared__ __bf16 ldsQ[4 * 2 * 512];       // [strip][kstep_d] A-frags
  __shared__ __bf16 ldsK[2][4 * 2 * 512];    // dbl-buf [ntile_keys][kstep_d] B-frags
  __shared__ __bf16 ldsV[2][4 * 2 * 512];    // dbl-buf [ntile_d][kstep_keys] B-frags
  __shared__ __bf16 ldsP[4 * 2 * 512];       // [wave][kstep_keys] A-frags

  const int tid = threadIdx.x, wave = tid >> 5, lane = tid & 31;
  const int q0 = blockIdx.x * 64;
  const int bh = blockIdx.y;
  const int b = bh >> 4, h = bh & 15;

  const __bf16* Q  = (const __bf16*)Qb  + (size_t)bh * SS * DH;
  const __bf16* K  = (const __bf16*)Kb  + (size_t)bh * SS * DH;
  const __bf16* Vt = (const __bf16*)Vtb + (size_t)bh * DH * SS;

  // Stage Q tile (64 x 64) once: b128 copies into A-fragment order.
  #pragma unroll
  for (int c = 0; c < 4; ++c) {
    int g = c * 128 + tid;                     // 0..511
    int m = g >> 3, dbase = (g & 7) * 8;
    int kstep = dbase >> 5, kwb = dbase & 31;
    int grp = kwb >> 4, hh = (kwb >> 3) & 1;
    unsigned off = (((m >> 4) << 1) + kstep) * 512 +
                   ((m & 15) + (hh << 4)) * 16 + grp * 8;
    *reinterpret_cast<uint4*>(&ldsQ[off]) =
        *reinterpret_cast<const uint4*>(&Q[(size_t)(q0 + m) * DH + dbase]);
  }

  // Issue async K/V fill of a tile into LDS buffer `buf` (8 asyncs / lane).
  auto issue_tile = [&](int kb0, int buf) {
    #pragma unroll
    for (int c = 0; c < 4; ++c) {
      int g = c * 128 + tid;
      // K: element (kl, d); fragment: contraction = d, n = local key
      int kl = g >> 3, dbase = (g & 7) * 8;
      int kstep = dbase >> 5, kwb = dbase & 31;
      int hh = (kwb >> 3) & 1, c8 = kwb >> 4;
      unsigned koff = (((kl >> 4) << 1) + kstep) * 512 +
                      ((kl & 15) + (hh << 4)) * 16 + c8 * 8;
      async_copy16(&ldsK[buf][koff], &K[(size_t)(kb0 + kl) * DH + dbase]);
      // V: element (d, kl); fragment: contraction = local key, n = d
      int d = g >> 3, klb = (g & 7) * 8;
      int vks = klb >> 5, vkw = klb & 31;
      int vhi = (vkw >> 3) & 1, veb = (vkw >> 4) * 8;
      unsigned voff = (((d >> 4) << 1) + vks) * 512 +
                      ((d & 15) + (vhi << 4)) * 16 + veb;
      async_copy16(&ldsV[buf][voff], &Vt[(size_t)d * SS + kb0 + klb]);
    }
  };

  issue_tile(0, 0);

  v16bf qf[2];
  // Q staged by plain ds stores; barrier below also covers it.
  float mstate[8], lstate[8];
  v8f o[4] = {};
  #pragma unroll
  for (int i = 0; i < 8; ++i) { mstate[i] = -3.0e38f; lstate[i] = 0.f; }

  const int hi = lane >> 4, nl = lane & 15;
  const int J = SS / 64;

  for (int j = 0; j < J; ++j) {
    const int kb0 = j * 64;
    const int cur = j & 1;

    if (j + 1 < J) {
      issue_tile(kb0 + 64, cur ^ 1);              // overlap next tile's DMA
      asm volatile("s_wait_asynccnt 0x8" ::: "memory");  // tile j complete
    } else {
      asm volatile("s_wait_asynccnt 0x0" ::: "memory");
    }
    __syncthreads();

    if (j == 0) {
      qf[0] = *reinterpret_cast<const v16bf*>(&ldsQ[(wave * 2 + 0) * 512]);
      qf[1] = *reinterpret_cast<const v16bf*>(&ldsQ[(wave * 2 + 1) * 512]);
    }
    // cache-hint upcoming bias rows (dominant HBM stream)
    __builtin_prefetch(&pos_bias[((size_t)h * SS + q0 + wave * 16) * SS + kb0 + 64], 0, 0);
    __builtin_prefetch(&mask[((size_t)b * SS + q0 + wave * 16) * SS + kb0 + 64], 0, 0);

    // scores tile 16 x 64 per wave (Q pre-scaled by 1/sqrt(DH))
    v8f sc[4] = {};
    #pragma unroll
    for (int t = 0; t < 4; ++t) {
      #pragma unroll
      for (int ks = 0; ks < 2; ++ks) {
        v16bf bf = *reinterpret_cast<const v16bf*>(&ldsK[cur][(t * 2 + ks) * 512]);
        sc[t] = __builtin_amdgcn_wmma_f32_16x16x32_bf16(
            false, qf[ks], false, bf, (short)0, sc[t], false, false);
      }
    }

    // position bias + mask
    #pragma unroll
    for (int t = 0; t < 4; ++t) {
      int key = kb0 + t * 16 + nl;
      #pragma unroll
      for (int i = 0; i < 8; ++i) {
        int q = q0 + wave * 16 + i + 8 * hi;
        float s = sc[t][i] + pos_bias[((size_t)h * SS + q) * SS + key];
        if (mask[((size_t)b * SS + q) * SS + key] == 0) s = -1.0e9f;
        sc[t][i] = s;
      }
    }

    // online softmax: each row lives in one 16-lane half at fixed VGPR index
    #pragma unroll
    for (int i = 0; i < 8; ++i) {
      float rm = fmaxf(fmaxf(sc[0][i], sc[1][i]), fmaxf(sc[2][i], sc[3][i]));
      #pragma unroll
      for (int off = 1; off < 16; off <<= 1)
        rm = fmaxf(rm, __shfl_xor(rm, off, 32));
      float mnew = fmaxf(mstate[i], rm);
      float alpha = __expf(mstate[i] - mnew);
      mstate[i] = mnew;
      float rs = 0.f;
      #pragma unroll
      for (int t = 0; t < 4; ++t) {
        float p = __expf(sc[t][i] - mnew);
        sc[t][i] = p;
        rs += p;
      }
      #pragma unroll
      for (int off = 1; off < 16; off <<= 1)
        rs += __shfl_xor(rs, off, 32);
      lstate[i] = lstate[i] * alpha + rs;
      #pragma unroll
      for (int t = 0; t < 4; ++t) o[t][i] = o[t][i] * alpha;
    }

    // P -> LDS bf16 A-fragments (contraction = 64 local keys, 2 ksteps)
    #pragma unroll
    for (int t = 0; t < 4; ++t) {
      int klocal = t * 16 + nl;
      int kstep = klocal >> 5, kw = klocal & 31;
      #pragma unroll
      for (int i = 0; i < 8; ++i) {
        int m = i + 8 * hi;
        ldsP[(wave * 2 + kstep) * 512 + a_frag_off(m, kw)] = (__bf16)sc[t][i];
      }
    }
    __syncthreads();

    // O += P @ V
    #pragma unroll
    for (int t = 0; t < 4; ++t) {
      #pragma unroll
      for (int ks = 0; ks < 2; ++ks) {
        v16bf pa = *reinterpret_cast<const v16bf*>(&ldsP[(wave * 2 + ks) * 512]);
        v16bf vb = *reinterpret_cast<const v16bf*>(&ldsV[cur][(t * 2 + ks) * 512]);
        o[t] = __builtin_amdgcn_wmma_f32_16x16x32_bf16(
            false, pa, false, vb, (short)0, o[t], false, false);
      }
    }
    __syncthreads();   // all waves done reading buffers before next overwrite
  }

  // normalize and write ctx in [B*S, DM] layout (head h -> cols h*64..)
  #pragma unroll
  for (int i = 0; i < 8; ++i) {
    int q = q0 + wave * 16 + i + 8 * hi;
    float inv = 1.0f / lstate[i];
    #pragma unroll
    for (int t = 0; t < 4; ++t) {
      int col = h * DH + t * 16 + nl;
      ctx[((size_t)b * SS + q) * DM + col] = o[t][i] * inv;
    }
  }
}

// ---------------------------------------------------------------------------
extern "C" void kernel_launch(void* const* d_in, const int* in_sizes, int n_in,
                              void* d_out, int out_size, void* d_ws, size_t ws_size,
                              hipStream_t stream) {
  const float* query = (const float*)d_in[0];
  const float* key_  = (const float*)d_in[1];
  const float* value = (const float*)d_in[2];
  const int*   mask  = (const int*)d_in[3];
  const float* pbias = (const float*)d_in[4];
  const float* w_q   = (const float*)d_in[5];
  const float* b_q   = (const float*)d_in[6];
  const float* w_k   = (const float*)d_in[7];
  const float* b_k   = (const float*)d_in[8];
  const float* w_v   = (const float*)d_in[9];
  const float* b_v   = (const float*)d_in[10];
  const float* w_o   = (const float*)d_in[11];
  const float* b_o   = (const float*)d_in[12];

  char* ws = (char*)d_ws;
  unsigned short* Qb  = (unsigned short*)(ws);                       // 8 MB bf16
  unsigned short* Kb  = (unsigned short*)(ws + ((size_t)8  << 20));  // 8 MB bf16
  unsigned short* Vtb = (unsigned short*)(ws + ((size_t)16 << 20));  // 8 MB bf16
  float*          cx  = (float*)        (ws + ((size_t)24 << 20));   // 16 MB f32

  const float qscale = 0.125f;   // 1/sqrt(DH=64), folded into Q projection
  dim3 blk(128);
  dim3 gp((BB * SS) / 64, DM / 64);

  proj_gemm_kernel<0><<<gp, blk, 0, stream>>>(query, w_q, b_q, Qb,  nullptr, qscale);
  proj_gemm_kernel<0><<<gp, blk, 0, stream>>>(key_,  w_k, b_k, Kb,  nullptr, 1.0f);
  proj_gemm_kernel<2><<<gp, blk, 0, stream>>>(value, w_v, b_v, Vtb, nullptr, 1.0f);

  dim3 ga(SS / 64, BB * NH);
  attention_kernel<<<ga, blk, 0, stream>>>(Qb, Kb, Vtb, pbias, mask, cx);

  proj_gemm_kernel<1><<<gp, blk, 0, stream>>>(cx, w_o, b_o, nullptr, (float*)d_out, 1.0f);
}
